// FluxJointAttention_90993177133185
// MI455X (gfx1250) — matrix operand order repaired and checked by reading, not compiled, not run on hardware
//
#include <hip/hip_runtime.h>

typedef __bf16 bf16;
typedef __attribute__((ext_vector_type(16))) __bf16 v16bf;
typedef __attribute__((ext_vector_type(8)))  __bf16 v8bf;
typedef __attribute__((ext_vector_type(4)))  __bf16 v4bf;
typedef __attribute__((ext_vector_type(8)))  float  v8f;
typedef __attribute__((ext_vector_type(4)))  unsigned int u32x4;
typedef __attribute__((ext_vector_type(8)))  int i32x8;
typedef __attribute__((ext_vector_type(4)))  int i32x4;

#define DIM      3072
#define HEADS    24
#define HEAD_DIM 128
#define IMG_LEN  2048
#define TXT_LEN  512
#define SEQ      (IMG_LEN + TXT_LEN)   // 2560

#if defined(__has_builtin)
#if __has_builtin(__builtin_amdgcn_tensor_load_to_lds) && \
    __has_builtin(__builtin_amdgcn_s_wait_tensorcnt)
#define USE_TDM 1
#endif
#if __has_builtin(__builtin_amdgcn_exp2f)
#define EXP2(x) __builtin_amdgcn_exp2f(x)
#else
#define EXP2(x) exp2f(x)
#endif
#else
#define EXP2(x) exp2f(x)
#endif

// Build a 16-element bf16 A/B fragment from two contiguous 8-element (16B) chunks.
static __device__ inline v16bf make_frag(const bf16* lo, const bf16* hi) {
  v8bf a = *(const v8bf*)lo;
  v8bf b = *(const v8bf*)hi;
  return __builtin_shufflevector(a, b, 0,1,2,3,4,5,6,7,8,9,10,11,12,13,14,15);
}

static __device__ inline v8f wmma_bf16(v16bf a, v16bf b, v8f c) {
  // 8 args: (neg_a, A, neg_b, B, c_mod, C, reuse_a, reuse_b)
  return __builtin_amdgcn_wmma_f32_16x16x32_bf16(false, a, false, b, (short)0, c,
                                                 false, false);
}

#ifdef USE_TDM
// Issue a TDM load of a [tileRows x 32] bf16 tile (row stride = rowStrideElems
// elements in global) into LDS at ldsAddr, padding each 64B row with 16B so
// LDS rows land at a 40-element stride (bank-conflict avoidance).
static __device__ inline void tdm_load_tile(const bf16* gptr, unsigned ldsAddr,
                                            unsigned tileRows,
                                            unsigned rowStrideElems) {
  const unsigned long long ga = (unsigned long long)(uintptr_t)gptr;
  u32x4 g0;
  g0[0] = 1u;                                   // count=1, user descriptor
  g0[1] = ldsAddr;                              // lds_addr (bytes)
  g0[2] = (unsigned)ga;                         // global_addr[31:0]
  g0[3] = (unsigned)(ga >> 32) | (2u << 30);    // global_addr[56:32] | type=2
  const unsigned td0 = 1u << 24;                // tensor dims huge: no OOB clamp
  const unsigned td1 = 1u << 24;
  i32x8 g1;
  g1[0] = (int)((1u << 16)      // data_size = 2 bytes
              | (1u << 20)      // pad_enable
              | (3u << 22)      // pad_interval: 16 DWORDs (one 64B tile row)
              | (3u << 25));    // pad_amount: 4 DWORDs (16B)
  g1[1] = (int)((td0 & 0xffffu) << 16);                    // tensor_dim0 lo16
  g1[2] = (int)((td0 >> 16) | ((td1 & 0xffffu) << 16));    // dim0 hi | dim1 lo
  g1[3] = (int)((td1 >> 16) | (32u << 16));                // dim1 hi | tile_dim0=32
  g1[4] = (int)tileRows;                                   // tile_dim1, tile_dim2=0
  g1[5] = (int)rowStrideElems;                             // tensor_dim0_stride
  g1[6] = 0;
  g1[7] = 0;
  const i32x4 z4 = {0, 0, 0, 0};
  const i32x8 z8 = {0, 0, 0, 0, 0, 0, 0, 0};
  __builtin_amdgcn_tensor_load_to_lds(g0, g1, z4, z4, z8, 0);
}
#endif

// ---------------------------------------------------------------------------
// Weight prep: Wt[n][k] = bf16(W[k][n])  (32x32 tiles through LDS)
// ---------------------------------------------------------------------------
__global__ __launch_bounds__(256) void transpose_convert_kernel(
    const float* __restrict__ W, bf16* __restrict__ Wt) {
  __shared__ float T[32][33];
  const int tx = threadIdx.x & 31;
  const int ty = threadIdx.x >> 5;      // 0..7
  const int n0 = blockIdx.x * 32;
  const int k0 = blockIdx.y * 32;
#pragma unroll
  for (int j = 0; j < 4; ++j)
    T[ty + 8 * j][tx] = W[(size_t)(k0 + ty + 8 * j) * DIM + n0 + tx];
  __syncthreads();
#pragma unroll
  for (int j = 0; j < 4; ++j)
    Wt[(size_t)(n0 + ty + 8 * j) * DIM + k0 + tx] = (bf16)T[tx][ty + 8 * j];
}

// Plain f32 -> bf16 (8 elements per thread)
__global__ __launch_bounds__(256) void convert_kernel(const float* __restrict__ X,
                                                      bf16* __restrict__ Y) {
  const size_t i = ((size_t)blockIdx.x * 256 + threadIdx.x) * 8;
  const float4 a = *(const float4*)(X + i);
  const float4 b = *(const float4*)(X + i + 4);
  v8bf o;
  o[0] = (bf16)a.x; o[1] = (bf16)a.y; o[2] = (bf16)a.z; o[3] = (bf16)a.w;
  o[4] = (bf16)b.x; o[5] = (bf16)b.y; o[6] = (bf16)b.z; o[7] = (bf16)b.w;
  *(v8bf*)(Y + i) = o;
}

// ---------------------------------------------------------------------------
// GEMM: C[M x 3072] = A[M x 3072](bf16) * Wt^T + bias. Wt is pre-transposed
// [n][k] bf16. Block tile 128x64, BK=32, 8 waves (4Mx2N), 32x32 per wave.
// Tiles staged via TDM (double-buffered) when available.
// ---------------------------------------------------------------------------
template <typename TOut>
__global__ __launch_bounds__(256) void gemm_kernel(const bf16* __restrict__ A,
                                                   const bf16* __restrict__ Wt,
                                                   const float* __restrict__ bias,
                                                   TOut* __restrict__ C) {
  __shared__ __align__(16) bf16 As[2][128][40];  // [buf][m][k], 40-elem rows
  __shared__ __align__(16) bf16 Bs[2][64][40];   // [buf][n][k]

  const int tid   = threadIdx.x;
  const int lane  = tid & 31;
  const int w     = tid >> 5;
  const int mBase = blockIdx.y * 128;
  const int nBase = blockIdx.x * 64;
  const int waveM = (w & 3) * 32;
  const int waveN = (w >> 2) * 32;
  const int l15   = lane & 15;
  const int hi16  = lane >> 4;

  v8f acc[2][2] = {};

#ifdef USE_TDM
  if (w == 0) {
    tdm_load_tile(A + (size_t)mBase * DIM, (unsigned)(uintptr_t)&As[0][0][0], 128, DIM);
    tdm_load_tile(Wt + (size_t)nBase * DIM, (unsigned)(uintptr_t)&Bs[0][0][0], 64, DIM);
  }
#endif

  for (int it = 0; it < DIM / 32; ++it) {
    const int kb = it * 32;
#ifdef USE_TDM
    const int buf = it & 1;
    if (w == 0) __builtin_amdgcn_s_wait_tensorcnt(0);  // tile `it` landed
    __syncthreads();  // tile visible to all; prev reads of other buffer done
    if (w == 0 && it + 1 < DIM / 32) {
      tdm_load_tile(A + (size_t)mBase * DIM + kb + 32,
                    (unsigned)(uintptr_t)&As[buf ^ 1][0][0], 128, DIM);
      tdm_load_tile(Wt + (size_t)nBase * DIM + kb + 32,
                    (unsigned)(uintptr_t)&Bs[buf ^ 1][0][0], 64, DIM);
    }
#else
    const int buf = 0;
    __syncthreads();
    {
      const int r0 = tid >> 2;          // 0..63
      const int c0 = (tid & 3) * 8;     // 0..24
      *(v8bf*)&As[0][r0][c0] =
          *(const v8bf*)(A + (size_t)(mBase + r0) * DIM + kb + c0);
      *(v8bf*)&As[0][64 + r0][c0] =
          *(const v8bf*)(A + (size_t)(mBase + 64 + r0) * DIM + kb + c0);
      *(v8bf*)&Bs[0][r0][c0] =
          *(const v8bf*)(Wt + (size_t)(nBase + r0) * DIM + kb + c0);
    }
    __syncthreads();
#endif

    // A 16x32 frag: lanes 0-15 K{0-7,16-23}, lanes 16-31 K{8-15,24-31}
    // B 32x16 frag: lanes 0-15 K{0-7,16-23}, lanes 16-31 K{8-15,24-31} (Bs is [n][k])
    const int aOff = hi16 ? 8 : 0;
    const int bOff = hi16 ? 8 : 0;
    v16bf af[2], bfr[2];
#pragma unroll
    for (int mi = 0; mi < 2; ++mi) {
      const bf16* base = &As[buf][waveM + mi * 16 + l15][0];
      af[mi] = make_frag(base + aOff, base + 16 + aOff);
    }
#pragma unroll
    for (int ni = 0; ni < 2; ++ni) {
      const bf16* base = &Bs[buf][waveN + ni * 16 + l15][0];
      bfr[ni] = make_frag(base + bOff, base + bOff + 16);
    }
#pragma unroll
    for (int mi = 0; mi < 2; ++mi)
#pragma unroll
      for (int ni = 0; ni < 2; ++ni)
        acc[mi][ni] = wmma_bf16(af[mi], bfr[ni], acc[mi][ni]);
  }

  // epilogue: bias + store (C layout: VGPR r -> row r + 8*hi16, col l15)
#pragma unroll
  for (int mi = 0; mi < 2; ++mi) {
#pragma unroll
    for (int ni = 0; ni < 2; ++ni) {
      const int col = nBase + waveN + ni * 16 + l15;
      const float b = bias[col];
#pragma unroll
      for (int r = 0; r < 8; ++r) {
        const int row = mBase + waveM + mi * 16 + r + 8 * hi16;
        C[(size_t)row * DIM + col] = (TOut)(acc[mi][ni][r] + b);
      }
    }
  }
}

// ---------------------------------------------------------------------------
// Fused per-head RMS-norm + RoPE (+ output scale), f32 in -> bf16 out.
// One wave per (token, head). For Q, oscale = log2(e)/sqrt(HEAD_DIM) so the
// attention kernel can skip score scaling and run softmax in base-2.
// ---------------------------------------------------------------------------
__global__ __launch_bounds__(256) void norm_rope_kernel(
    const float* __restrict__ X, bf16* __restrict__ Y,
    const float* __restrict__ g_txt, const float* __restrict__ g_img,
    const float* __restrict__ cosT, const float* __restrict__ sinT,
    float oscale) {
  const int lane  = threadIdx.x & 31;
  const int w     = threadIdx.x >> 5;
  const int pair  = blockIdx.x * 8 + w;     // token*HEADS + head
  const int token = pair / HEADS;
  const int head  = pair % HEADS;
  const int d0    = lane * 4;

  const float4 x = *(const float4*)(X + (size_t)token * DIM + head * HEAD_DIM + d0);
  float ss = x.x * x.x + x.y * x.y + x.z * x.z + x.w * x.w;
#pragma unroll
  for (int off = 1; off < 32; off <<= 1) ss += __shfl_xor(ss, off, 32);
  const float r = rsqrtf(ss * (1.0f / HEAD_DIM) + 1e-6f);

  const float* g = (token < TXT_LEN) ? g_txt : g_img;
  const float4 gv = *(const float4*)(g + d0);
  const float4 cv = *(const float4*)(cosT + (size_t)token * HEAD_DIM + d0);
  const float4 sv = *(const float4*)(sinT + (size_t)token * HEAD_DIM + d0);

  const float y0 = x.x * r * gv.x, y1 = x.y * r * gv.y;
  const float y2 = x.z * r * gv.z, y3 = x.w * r * gv.w;
  // out[2i] = y[2i]*cos - y[2i+1]*sin ; out[2i+1] = y[2i+1]*cos + y[2i]*sin
  v4bf o;
  o[0] = (bf16)((y0 * cv.x - y1 * sv.x) * oscale);
  o[1] = (bf16)((y1 * cv.y + y0 * sv.y) * oscale);
  o[2] = (bf16)((y2 * cv.z - y3 * sv.z) * oscale);
  o[3] = (bf16)((y3 * cv.w + y2 * sv.w) * oscale);
  *(v4bf*)(Y + (size_t)token * DIM + head * HEAD_DIM + d0) = o;
}

// ---------------------------------------------------------------------------
// Flash attention: grid (SEQ/128, HEADS), 8 waves x 16 queries, key blocks of 32.
// Scores arrive pre-scaled by log2(e)/sqrt(D); softmax runs in base-2 (exp2).
// ---------------------------------------------------------------------------
__global__ __launch_bounds__(256) void attn_kernel(const bf16* __restrict__ Q,
                                                   const bf16* __restrict__ K,
                                                   const bf16* __restrict__ V,
                                                   bf16* __restrict__ O) {
  __shared__ __align__(16) bf16 Vl[HEAD_DIM][40];  // V transposed: [dim][key]
  __shared__ __align__(16) bf16 Pl[8][16][32];     // per-wave P scratch

  const int tid   = threadIdx.x;
  const int lane  = tid & 31;
  const int w     = tid >> 5;
  const int h     = blockIdx.y;
  const int qBase = blockIdx.x * 128 + w * 16;
  const int l15   = lane & 15;
  const int hi16  = lane >> 4;

  // Q held in registers as 4 A-fragments (16 queries x 128 dims)
  v16bf qf[4];
  {
    const bf16* qrow = Q + (size_t)(qBase + l15) * DIM + h * HEAD_DIM;
    const int aOff = hi16 ? 8 : 0;
#pragma unroll
    for (int c = 0; c < 4; ++c)
      qf[c] = make_frag(qrow + c * 32 + aOff, qrow + c * 32 + 16 + aOff);
  }

  v8f o[8] = {};
  float mrow[8], srow[8];
#pragma unroll
  for (int r = 0; r < 8; ++r) { mrow[r] = -__builtin_inff(); srow[r] = 0.0f; }

  for (int kb = 0; kb < SEQ; kb += 32) {
    __syncthreads();
    // stage V block transposed: Vl[dim][key]
    {
      const int key = tid >> 3;        // 0..31
      const int d0  = (tid & 7) * 16;  // 0..112
      const bf16* vp = V + (size_t)(kb + key) * DIM + h * HEAD_DIM + d0;
      v8bf a = *(const v8bf*)vp;
      v8bf b = *(const v8bf*)(vp + 8);
#pragma unroll
      for (int i = 0; i < 8; ++i) { Vl[d0 + i][key] = a[i]; Vl[d0 + 8 + i][key] = b[i]; }
    }
    __syncthreads();

    // S = Q K^T for 32 keys (two 16x16 tiles), K^T B-frags gathered from global
    v8f s0 = {}, s1 = {};
    {
      const int bOff = hi16 ? 8 : 0;
      const bf16* k0row = K + (size_t)(kb + l15) * DIM + h * HEAD_DIM;
      const bf16* k1row = K + (size_t)(kb + 16 + l15) * DIM + h * HEAD_DIM;
#pragma unroll
      for (int c = 0; c < 4; ++c) {
        v16bf b0 = make_frag(k0row + c * 32 + bOff, k0row + c * 32 + bOff + 16);
        v16bf b1 = make_frag(k1row + c * 32 + bOff, k1row + c * 32 + bOff + 16);
        s0 = wmma_bf16(qf[c], b0, s0);
        s1 = wmma_bf16(qf[c], b1, s1);
      }
    }

    // base-2 online softmax (row = r + 8*hi16; row spans 16 lanes of a half)
    float alpha[8];
#pragma unroll
    for (int r = 0; r < 8; ++r) {
      const float t0 = s0[r], t1 = s1[r];
      float rm = fmaxf(t0, t1);
#pragma unroll
      for (int off = 1; off < 16; off <<= 1) rm = fmaxf(rm, __shfl_xor(rm, off, 32));
      const float mn = fmaxf(mrow[r], rm);
      alpha[r] = EXP2(mrow[r] - mn);
      mrow[r] = mn;
      const float p0 = EXP2(t0 - mn);
      const float p1 = EXP2(t1 - mn);
      s0[r] = p0; s1[r] = p1;
      float rs = p0 + p1;
#pragma unroll
      for (int off = 1; off < 16; off <<= 1) rs += __shfl_xor(rs, off, 32);
      srow[r] = srow[r] * alpha[r] + rs;
    }
#pragma unroll
    for (int f = 0; f < 8; ++f)
#pragma unroll
      for (int r = 0; r < 8; ++r) o[f][r] *= alpha[r];

    // P: C-layout -> LDS -> reload in A-fragment layout (cross-lane, same wave)
#pragma unroll
    for (int r = 0; r < 8; ++r) {
      Pl[w][r + 8 * hi16][l15]      = (bf16)s0[r];
      Pl[w][r + 8 * hi16][16 + l15] = (bf16)s1[r];
    }
    asm volatile("s_wait_dscnt 0x0" ::: "memory");
    v16bf pf;
    {
      const int aOff = hi16 ? 8 : 0;
      const bf16* prow = &Pl[w][l15][0];
      pf = make_frag(prow + aOff, prow + 16 + aOff);
    }

    // O += P V : B-frags from transposed V in LDS
#pragma unroll
    for (int f = 0; f < 8; ++f) {
      const bf16* vrow = &Vl[f * 16 + l15][0];
      const int bOff = hi16 ? 8 : 0;
      v16bf vf = make_frag(vrow + bOff, vrow + bOff + 16);
      o[f] = wmma_bf16(pf, vf, o[f]);
    }
  }

  // normalize and store
  float inv[8];
#pragma unroll
  for (int r = 0; r < 8; ++r) inv[r] = 1.0f / srow[r];
#pragma unroll
  for (int f = 0; f < 8; ++f) {
#pragma unroll
    for (int r = 0; r < 8; ++r) {
      const int token = qBase + r + 8 * hi16;
      const int d = f * 16 + l15;
      O[(size_t)token * DIM + h * HEAD_DIM + d] = (bf16)(o[f][r] * inv[r]);
    }
  }
}

// ---------------------------------------------------------------------------
extern "C" void kernel_launch(void* const* d_in, const int* in_sizes, int n_in,
                              void* d_out, int out_size, void* d_ws, size_t ws_size,
                              hipStream_t stream) {
  (void)in_sizes; (void)n_in; (void)out_size; (void)ws_size;
  const float* hidden = (const float*)d_in[0];
  const float* enc    = (const float*)d_in[1];
  const float* rcos   = (const float*)d_in[2];
  const float* rsin   = (const float*)d_in[3];
  const float* Wf[8]  = {(const float*)d_in[4],  (const float*)d_in[5],
                         (const float*)d_in[6],  (const float*)d_in[7],
                         (const float*)d_in[8],  (const float*)d_in[9],
                         (const float*)d_in[10], (const float*)d_in[11]};
  // order: Wq Wk Wv Waq Wak Wav Wout Wadd
  const float* bq   = (const float*)d_in[12];
  const float* bk   = (const float*)d_in[13];
  const float* bv   = (const float*)d_in[14];
  const float* baq  = (const float*)d_in[15];
  const float* bak  = (const float*)d_in[16];
  const float* bav  = (const float*)d_in[17];
  const float* boutp= (const float*)d_in[18];
  const float* badd = (const float*)d_in[19];
  const float* g_q  = (const float*)d_in[20];
  const float* g_k  = (const float*)d_in[21];
  const float* g_aq = (const float*)d_in[22];
  const float* g_ak = (const float*)d_in[23];
  float* out = (float*)d_out;

  char* ws = (char*)d_ws;
  const size_t SEQD = (size_t)SEQ * DIM;
  const size_t WD   = (size_t)DIM * DIM;
  float* Qf = (float*)ws; ws += SEQD * sizeof(float);
  float* Kf = (float*)ws; ws += SEQD * sizeof(float);
  bf16* Qb = (bf16*)ws;  ws += SEQD * sizeof(bf16);
  bf16* Kb = (bf16*)ws;  ws += SEQD * sizeof(bf16);
  bf16* Vb = (bf16*)ws;  ws += SEQD * sizeof(bf16);
  bf16* Ab = (bf16*)ws;  ws += SEQD * sizeof(bf16);
  bf16* Xb = (bf16*)ws;  ws += SEQD * sizeof(bf16);   // [txt | img] activations
  bf16* Wt[8];
  for (int i = 0; i < 8; ++i) { Wt[i] = (bf16*)ws; ws += WD * sizeof(bf16); }

  const dim3 blk(256);
  const size_t txtOff = (size_t)TXT_LEN * DIM;  // joint order: [txt | img]

  // ---- precision prep: weights transposed->bf16, activations ->bf16
  const dim3 gTr(DIM / 32, DIM / 32);
  const dim3 bTr(256);
  for (int i = 0; i < 8; ++i)
    transpose_convert_kernel<<<gTr, bTr, 0, stream>>>(Wf[i], Wt[i]);
  convert_kernel<<<dim3((TXT_LEN * DIM) / 2048), blk, 0, stream>>>(enc, Xb);
  convert_kernel<<<dim3((IMG_LEN * DIM) / 2048), blk, 0, stream>>>(hidden, Xb + txtOff);

  const dim3 gImg(DIM / 64, IMG_LEN / 128);
  const dim3 gTxt(DIM / 64, TXT_LEN / 128);

  // QKV projections (txt rows 0..511, img rows 512..2559 of joint buffers)
  gemm_kernel<float><<<gImg, blk, 0, stream>>>(Xb + txtOff, Wt[0], bq,  Qf + txtOff);
  gemm_kernel<float><<<gTxt, blk, 0, stream>>>(Xb,          Wt[3], baq, Qf);
  gemm_kernel<float><<<gImg, blk, 0, stream>>>(Xb + txtOff, Wt[1], bk,  Kf + txtOff);
  gemm_kernel<float><<<gTxt, blk, 0, stream>>>(Xb,          Wt[4], bak, Kf);
  gemm_kernel<bf16 ><<<gImg, blk, 0, stream>>>(Xb + txtOff, Wt[2], bv,  Vb + txtOff);
  gemm_kernel<bf16 ><<<gTxt, blk, 0, stream>>>(Xb,          Wt[5], bav, Vb);

  // per-head RMS norm + RoPE -> bf16; fold log2(e)/sqrt(D) into Q
  const float qscale = 1.44269504088896340f * 0.08838834764831845f;
  const dim3 gNR((SEQ * HEADS) / 8);
  norm_rope_kernel<<<gNR, blk, 0, stream>>>(Qf, Qb, g_aq, g_q, rcos, rsin, qscale);
  norm_rope_kernel<<<gNR, blk, 0, stream>>>(Kf, Kb, g_ak, g_k, rcos, rsin, 1.0f);

  // attention
  const dim3 gAtt(SEQ / 128, HEADS);
  attn_kernel<<<gAtt, blk, 0, stream>>>(Qb, Kb, Vb, Ab);

  // output projections: img -> out[0:2048*3072], txt -> out[2048*3072:]
  gemm_kernel<float><<<gImg, blk, 0, stream>>>(Ab + txtOff, Wt[6], boutp, out);
  gemm_kernel<float><<<gTxt, blk, 0, stream>>>(Ab, Wt[7], badd,
                                               out + (size_t)IMG_LEN * DIM);
}
